// DistributedMoE_57732950393352
// MI455X (gfx1250) — matrix operand order repaired and checked by reading, not compiled, not run on hardware
//
#include <hip/hip_runtime.h>
#include <hip/hip_bf16.h>

#define BB 4096
#define HH 1024
#define DD 4096
#define EE 8
#define MT 32          // tokens per tile (two 16-row m-tiles)
#define DC 128         // D-chunk width (8 waves x 16)
#define TILES_MAX (BB / MT)   // 128

typedef __attribute__((ext_vector_type(16))) __bf16 v16bf;
typedef __attribute__((ext_vector_type(8)))  float  v8f;

union BF16x16 { v16bf v; unsigned short u[16]; uint4 q[2]; };

__device__ __forceinline__ unsigned short f2bf(float f) {
  unsigned int u = __float_as_uint(f);
  unsigned int r = (u + 0x7FFFu + ((u >> 16) & 1u)) >> 16;  // round-to-nearest-even
  return (unsigned short)r;
}

// ---------------- Gating: logits -> softmax -> top-2 -> compaction ----------------
__global__ __launch_bounds__(256) void gate_kernel(
    const float* __restrict__ x, const float* __restrict__ gw,
    const float* __restrict__ gb,
    int* cnt, float* sumP, float* sumR, int* etok, float* escale) {
  int wave = threadIdx.x >> 5;
  int lane = threadIdx.x & 31;
  int b = blockIdx.x * 8 + wave;
  if (b >= BB) return;

  float acc[EE];
#pragma unroll
  for (int e = 0; e < EE; ++e) acc[e] = 0.f;
  const float* xr = x + (size_t)b * HH;
  for (int h = lane; h < HH; h += 32) {
    float xv = xr[h];
    const float* g = gw + (size_t)h * EE;
#pragma unroll
    for (int e = 0; e < EE; ++e) acc[e] += xv * g[e];
  }
#pragma unroll
  for (int off = 16; off >= 1; off >>= 1) {
#pragma unroll
    for (int e = 0; e < EE; ++e) acc[e] += __shfl_xor(acc[e], off, 32);
  }
  if (lane == 0) {
    float p[EE];
    float mx = -1e30f;
#pragma unroll
    for (int e = 0; e < EE; ++e) { p[e] = acc[e] + gb[e]; mx = fmaxf(mx, p[e]); }
    float s = 0.f;
#pragma unroll
    for (int e = 0; e < EE; ++e) { p[e] = __expf(p[e] - mx); s += p[e]; }
    float inv = 1.f / s;
    int i1 = 0, i2 = 0; float p1 = -1.f, p2 = -2.f;
#pragma unroll
    for (int e = 0; e < EE; ++e) {
      p[e] *= inv;
      atomicAdd(&sumP[e], p[e]);
      if (p[e] > p1)      { p2 = p1; i2 = i1; p1 = p[e]; i1 = e; }
      else if (p[e] > p2) { p2 = p[e]; i2 = e; }
    }
    int pos1 = atomicAdd(&cnt[i1], 1);
    etok[i1 * BB + pos1] = b;  escale[i1 * BB + pos1] = p1 * 0.5f;
    int pos2 = atomicAdd(&cnt[i2], 1);
    etok[i2 * BB + pos2] = b;  escale[i2 * BB + pos2] = p2 * 0.5f;
    atomicAdd(&sumR[i1], 1.f);
    atomicAdd(&sumR[i2], 1.f);
  }
}

// ---------------- Aux loss finalize ----------------
__global__ void aux_kernel(const float* sumP, const float* sumR, float* out_aux) {
  if (threadIdx.x == 0) {
    float a = 0.f;
#pragma unroll
    for (int e = 0; e < EE; ++e)
      a += (sumP[e] * (1.f / BB)) * (sumR[e] * (1.f / BB));
    *out_aux = a * (float)EE;
  }
}

// ------------- Transpose+convert: E x [R][C] fp32  ->  E x [C][R] bf16 -------------
__global__ __launch_bounds__(256) void transcvt_kernel(
    const float* __restrict__ src, unsigned short* __restrict__ dst,
    int R, int C) {
  __shared__ float tile[32][33];
  int e = blockIdx.z;
  const float* s = src + (size_t)e * R * C;
  unsigned short* d = dst + (size_t)e * R * C;
  int c0 = blockIdx.x * 32, r0 = blockIdx.y * 32;
  int tx = threadIdx.x & 31, ty = threadIdx.x >> 5;   // 32 x 8
#pragma unroll
  for (int i = 0; i < 32; i += 8)
    tile[ty + i][tx] = s[(size_t)(r0 + ty + i) * C + c0 + tx];
  __syncthreads();
#pragma unroll
  for (int i = 0; i < 32; i += 8)
    d[(size_t)(c0 + ty + i) * R + r0 + tx] = f2bf(tile[tx][ty + i]);
}

// ---------------- Expert MLP: fused relu(x@w1+b1)@w2+b2, scaled scatter ----------------
// PRE=true: weights pre-transposed to bf16 (w1t[E][D][H], w2t[E][H][D]) -> b128 loads.
// MT=32: each B fragment is reused by two WMMAs (m-tiles 0/1) -> halved weight traffic.
template <bool PRE>
__global__ __launch_bounds__(256) void moe_kernel(
    const float* __restrict__ x,
    const float* __restrict__ w1, const float* __restrict__ b1,
    const float* __restrict__ w2, const float* __restrict__ b2,
    const unsigned short* __restrict__ w1t, const unsigned short* __restrict__ w2t,
    const int* __restrict__ cnt, const int* __restrict__ etok,
    const float* __restrict__ escale, float* __restrict__ y) {

  __shared__ __align__(16) unsigned short xs[MT][HH + 8];   // 32x1032 bf16 = 66 KB
  __shared__ __align__(16) unsigned short hs[MT][DC + 8];   // 32x136  bf16 = 8.7 KB
  __shared__ int   stok[MT];
  __shared__ float sscale[MT];

  int e    = blockIdx.x / TILES_MAX;          // expert-major: weights stay hot in L2
  int tile = blockIdx.x % TILES_MAX;
  int ntok = cnt[e];
  if (tile * MT >= ntok) return;              // block-uniform early exit

  int tid = threadIdx.x;
  if (tid < MT) {
    int idx = tile * MT + tid;
    if (idx < ntok) { stok[tid] = etok[e * BB + idx]; sscale[tid] = escale[e * BB + idx]; }
    else            { stok[tid] = 0;                  sscale[tid] = 0.f; }
  }
  __syncthreads();

  // gather MT token rows of x into LDS as bf16 (vectorized: b128 in, b64 out)
  const int HH4 = HH / 4;
  for (int i = tid; i < MT * HH4; i += 256) {
    int m = i / HH4, h = (i % HH4) * 4;
    float4 v = *(const float4*)(x + (size_t)stok[m] * HH + h);
    union { unsigned short u[4]; uint2 d; } o;
    o.u[0] = f2bf(v.x); o.u[1] = f2bf(v.y); o.u[2] = f2bf(v.z); o.u[3] = f2bf(v.w);
    *(uint2*)&xs[m][h] = o.d;
  }
  __syncthreads();

  int wave  = tid >> 5;
  int lane  = tid & 31;
  int mrow  = lane & 15;           // A-matrix row / B-matrix column
  int ahalf = (lane >> 4) * 16;    // A frag: K 0-15 (lo lanes) vs 16-31 (hi lanes)
  int bhalf = (lane >> 4) * 16;    // B frag: K 0-15 (lo lanes) vs 16-31 (hi lanes)
  int mhalf = (lane >> 4) << 3;    // C/D frag: rows 0-7 (lo lanes) vs 8-15 (hi lanes)

  const float* w1e = w1 + (size_t)e * HH * DD;
  const float* w2e = w2 + (size_t)e * DD * HH;
  const unsigned short* w1te = w1t + (size_t)e * DD * HH;   // [D][H] bf16
  const unsigned short* w2te = w2t + (size_t)e * HH * DD;   // [H][D] bf16

  v8f zero = {};
  v8f out_acc[16];                 // [mt][t] : mt*8 + t
#pragma unroll
  for (int t = 0; t < 16; ++t) out_acc[t] = zero;

  for (int dc = 0; dc < DD; dc += DC) {
    // ---- phase A: h[:, dc + wave*16 .. +15] = relu(x @ w1 + b1) ----
    v8f hacc0 = zero, hacc1 = zero;
    int ncol = dc + wave * 16 + mrow;
    for (int ks = 0; ks < HH; ks += 32) {
      BF16x16 A0, A1, Bm;
      A0.q[0] = *(const uint4*)&xs[mrow][ks + ahalf];
      A0.q[1] = *(const uint4*)&xs[mrow][ks + 8 + ahalf];
      A1.q[0] = *(const uint4*)&xs[16 + mrow][ks + ahalf];
      A1.q[1] = *(const uint4*)&xs[16 + mrow][ks + 8 + ahalf];
      if (PRE) {
        const unsigned short* bp = w1te + (size_t)ncol * HH + ks + bhalf;
        Bm.q[0] = *(const uint4*)(bp);
        Bm.q[1] = *(const uint4*)(bp + 8);
      } else {
#pragma unroll
        for (int j = 0; j < 16; ++j)
          Bm.u[j] = f2bf(w1e[(size_t)(ks + bhalf + j) * DD + ncol]);
      }
      hacc0 = __builtin_amdgcn_wmma_f32_16x16x32_bf16(
          false, A0.v, false, Bm.v, (short)0, hacc0, false, false);
      hacc1 = __builtin_amdgcn_wmma_f32_16x16x32_bf16(
          false, A1.v, false, Bm.v, (short)0, hacc1, false, false);
    }
    {
      int nloc = wave * 16 + mrow;
      float bias = b1[(size_t)e * DD + dc + nloc];
#pragma unroll
      for (int r = 0; r < 8; ++r) {
        int m = r + mhalf;
        hs[m][nloc]      = f2bf(fmaxf(hacc0[r] + bias, 0.f));
        hs[16 + m][nloc] = f2bf(fmaxf(hacc1[r] + bias, 0.f));
      }
    }
    __syncthreads();

    // ---- phase B: out[:, wave*128 + t*16] += h_chunk @ w2[dc-chunk, :] ----
#pragma unroll
    for (int kk = 0; kk < 4; ++kk) {
      BF16x16 A0, A1;
      int kb = kk * 32;
      A0.q[0] = *(const uint4*)&hs[mrow][kb + ahalf];
      A0.q[1] = *(const uint4*)&hs[mrow][kb + 8 + ahalf];
      A1.q[0] = *(const uint4*)&hs[16 + mrow][kb + ahalf];
      A1.q[1] = *(const uint4*)&hs[16 + mrow][kb + 8 + ahalf];
#pragma unroll
      for (int t = 0; t < 8; ++t) {
        BF16x16 Bm;
        int col = wave * 128 + t * 16 + mrow;
        if (PRE) {
          const unsigned short* bp = w2te + (size_t)col * DD + dc + kb + bhalf;
          Bm.q[0] = *(const uint4*)(bp);
          Bm.q[1] = *(const uint4*)(bp + 8);
        } else {
#pragma unroll
          for (int j = 0; j < 16; ++j)
            Bm.u[j] = f2bf(w2e[(size_t)(dc + kb + bhalf + j) * HH + col]);
        }
        out_acc[t] = __builtin_amdgcn_wmma_f32_16x16x32_bf16(
            false, A0.v, false, Bm.v, (short)0, out_acc[t], false, false);
        out_acc[8 + t] = __builtin_amdgcn_wmma_f32_16x16x32_bf16(
            false, A1.v, false, Bm.v, (short)0, out_acc[8 + t], false, false);
      }
    }
    __syncthreads();
  }

  // ---- epilogue: +b2, scale by routing prob, scatter-accumulate into y ----
#pragma unroll
  for (int t = 0; t < 8; ++t) {
    int col = wave * 128 + t * 16 + mrow;
    float bias = b2[(size_t)e * HH + col];
#pragma unroll
    for (int r = 0; r < 8; ++r) {
      int m = r + mhalf;
      float v0 = (out_acc[t][r] + bias) * sscale[m];
      atomicAdd(&y[(size_t)stok[m] * HH + col], v0);
      float v1 = (out_acc[8 + t][r] + bias) * sscale[16 + m];
      atomicAdd(&y[(size_t)stok[16 + m] * HH + col], v1);
    }
  }
}

extern "C" void kernel_launch(void* const* d_in, const int* in_sizes, int n_in,
                              void* d_out, int out_size, void* d_ws, size_t ws_size,
                              hipStream_t stream) {
  const float* x  = (const float*)d_in[0];
  const float* gw = (const float*)d_in[1];
  const float* gb = (const float*)d_in[2];
  const float* w1 = (const float*)d_in[3];
  const float* b1 = (const float*)d_in[4];
  const float* w2 = (const float*)d_in[5];
  const float* b2 = (const float*)d_in[6];
  float* y = (float*)d_out;

  char* ws      = (char*)d_ws;
  int*   cnt    = (int*)  ws;                       // [E]
  float* sumP   = (float*)(ws + 32);                // [E]
  float* sumR   = (float*)(ws + 64);                // [E]
  int*   etok   = (int*)  (ws + 96);                // [E*B]
  float* escale = (float*)(ws + 96 + (size_t)EE * BB * 4); // [E*B]

  size_t w1t_off = 96 + 2 * (size_t)EE * BB * 4;
  size_t wmat    = (size_t)EE * DD * HH * 2;        // 64 MB each (bf16)
  size_t w2t_off = w1t_off + wmat;
  size_t need    = w2t_off + wmat;
  unsigned short* w1t = (unsigned short*)(ws + w1t_off);
  unsigned short* w2t = (unsigned short*)(ws + w2t_off);
  bool pre = (ws_size >= need);                     // host-side, deterministic

  hipMemsetAsync(ws, 0, 96, stream);
  hipMemsetAsync(d_out, 0, (size_t)BB * HH * sizeof(float), stream);

  gate_kernel<<<BB / 8, 256, 0, stream>>>(x, gw, gb, cnt, sumP, sumR, etok, escale);
  aux_kernel<<<1, 32, 0, stream>>>(sumP, sumR, y + (size_t)BB * HH);

  if (pre) {
    // w1[E][H][D] -> w1t[E][D][H] bf16 ; w2[E][D][H] -> w2t[E][H][D] bf16
    transcvt_kernel<<<dim3(DD / 32, HH / 32, EE), 256, 0, stream>>>(w1, w1t, HH, DD);
    transcvt_kernel<<<dim3(HH / 32, DD / 32, EE), 256, 0, stream>>>(w2, w2t, DD, HH);
    moe_kernel<true><<<EE * TILES_MAX, 256, 0, stream>>>(
        x, w1, b1, w2, b2, w1t, w2t, cnt, etok, escale, y);
  } else {
    moe_kernel<false><<<EE * TILES_MAX, 256, 0, stream>>>(
        x, w1, b1, w2, b2, w1t, w2t, cnt, etok, escale, y);
  }
}
